// Encoder_3667902071346
// MI455X (gfx1250) — matrix operand order repaired and checked by reading, not compiled, not run on hardware
//
#include <hip/hip_runtime.h>

// Problem constants (from reference): B,T,V,E,H,L = 512,128,128,512,512,2
constexpr int NB = 512;      // batch
constexpr int NT = 128;      // seq len
constexpr int NV = 128;      // vocab
constexpr int NE = 512;      // embed dim
constexpr int NH = 512;      // hidden dim
constexpr int NL = 2;        // layers
constexpr int NG = 3 * NH;   // 1536 gate width

constexpr int MBLK   = 128;            // per-block M (8 waves x 16)
constexpr int NBLK   = 64;             // per-block N (4 WMMA tiles, shared B)
constexpr int KSTEP  = 32;             // WMMA K
constexpr int LDSCOL = 40;             // ushorts per staged column (32 + 8 pad -> 80B stride)
constexpr int LDSBUF = NBLK * LDSCOL;  // ushorts per LDS buffer

typedef __attribute__((ext_vector_type(16))) __bf16        v16bf;
typedef __attribute__((ext_vector_type(8)))  float         v8f;
typedef __attribute__((ext_vector_type(8)))  unsigned int  v8u;

// ---- CDNA5 async global->LDS path (guarded; sync ds_store fallback) --------
#if defined(__HIP_DEVICE_COMPILE__) && defined(__has_builtin)
#if __has_builtin(__builtin_amdgcn_global_load_async_to_lds_b128)
#define USE_ASYNC_LDS 1
#endif
#endif

#ifdef USE_ASYNC_LDS
// builtin signature (from clang diagnostic): (int4 AS1*, int4 AS3*, Imm int, Imm int)
typedef int v4i_gcc __attribute__((vector_size(16)));
typedef __attribute__((address_space(1))) v4i_gcc* g_v4i_ptr;
typedef __attribute__((address_space(3))) v4i_gcc* l_v4i_ptr;
#define GPTR(p) ((g_v4i_ptr)(p))
#define LPTR(p) ((l_v4i_ptr)(p))
__device__ __forceinline__ void wait_async0() {
#if __has_builtin(__builtin_amdgcn_s_wait_asynccnt)
    __builtin_amdgcn_s_wait_asynccnt(0);
#else
    asm volatile("s_wait_asynccnt 0" ::: "memory");
#endif
}
#endif

__device__ __forceinline__ unsigned short f32_to_bf16(float f) {
    unsigned int u = __float_as_uint(f);
    unsigned int r = 0x7FFFu + ((u >> 16) & 1u);  // round-to-nearest-even
    return (unsigned short)((u + r) >> 16);
}

__device__ __forceinline__ float sigmoidf(float x) {
    return 1.0f / (1.0f + __expf(-x));
}

__device__ __forceinline__ v16bf as_bf16(v8u v) { return __builtin_bit_cast(v16bf, v); }

// ---------------------------------------------------------------------------
// f32 -> bf16 conversion
// ---------------------------------------------------------------------------
__global__ void cvt_f32_bf16(const float* __restrict__ src,
                             unsigned short* __restrict__ dst, int n) {
    int i = blockIdx.x * blockDim.x + threadIdx.x;
    if (i < n) dst[i] = f32_to_bf16(src[i]);
}

// ---------------------------------------------------------------------------
// Block-tiled WMMA GEMM body:
//   C[M,N] (f32) = A[M,K] (bf16, row-major) * W[N,K]^T (bf16, row-major)
// Block = 256 threads = 8 waves, tile 128(M) x 64(N). All 8 waves share the
// same B slice, double-buffered in LDS (async global->LDS when available).
// Next-step A fragment is register-prefetched, so WMMA never waits on a load
// issued in the same iteration.
// ---------------------------------------------------------------------------
__device__ __forceinline__ void gemm_tile(
    const unsigned short* __restrict__ A,   // [M,K] bf16
    const unsigned short* __restrict__ W,   // [N,K] bf16
    float* __restrict__ C,                  // [M,N] f32
    int M, int N, int K,
    unsigned short* Bsh)                    // LDS: 2 * LDSBUF ushorts
{
    const int tid  = threadIdx.x;
    const int lane = tid & 31;
    const int wave = tid >> 5;

    const int tilesN = N / NBLK;
    const int bm = (int)blockIdx.x / tilesN;
    const int bn = (int)blockIdx.x % tilesN;
    const int m0 = bm * MBLK + wave * 16;
    const int n0 = bn * NBLK;
    if (m0 >= M) return;

    const int khalf = lane >> 4;                 // K-half select (contiguous K order, same for A and B)
    const unsigned short* Arow = A + (size_t)(m0 + (lane & 15)) * K;

    // staging: each of 256 threads copies 16B of W per K-step
    const int scol   = tid >> 2;                 // 0..63 (column within N-slice)
    const int spart  = (tid & 3) * 8;            // ushort offset within the 32-K row
    const unsigned short* Wst = W + (size_t)(n0 + scol) * K + spart;
    const int ldsOff = scol * LDSCOL + spart;

    v8f acc[4] = {v8f{}, v8f{}, v8f{}, v8f{}};

    auto loadA = [&](int k0) -> v8u {
        // A 16x32: lanes 0-15 K=0..15, lanes 16-31 K=16..31 (contiguous, matches B)
        const uint4 lo = *(const uint4*)(Arow + k0 + khalf * 16);
        const uint4 hi = *(const uint4*)(Arow + k0 + khalf * 16 + 8);
        v8u r = {lo.x, lo.y, lo.z, lo.w, hi.x, hi.y, hi.z, hi.w};
        return r;
    };

    // ---- prime pipeline: stage K-step 0 into buffer 0
#ifdef USE_ASYNC_LDS
    __builtin_amdgcn_global_load_async_to_lds_b128(GPTR(Wst), LPTR(Bsh + ldsOff), 0, 0);
#else
    *(uint4*)(Bsh + ldsOff) = *(const uint4*)(Wst);
#endif
    v8u aReg = loadA(0);
#ifdef USE_ASYNC_LDS
    wait_async0();
#endif
    __syncthreads();

    const int nsteps = K / KSTEP;
    for (int s = 0; s < nsteps; ++s) {
        const int buf = s & 1;
        const int k0  = s * KSTEP;
        const bool more = (s + 1) < nsteps;

        v8u  aNext = {};
        uint4 wNext = {};
        if (more) {
            // kick off next B slice first (overlaps with WMMA below)
#ifdef USE_ASYNC_LDS
            __builtin_amdgcn_global_load_async_to_lds_b128(
                GPTR(Wst + k0 + KSTEP),
                LPTR(Bsh + (buf ^ 1) * LDSBUF + ldsOff), 0, 0);
#else
            wNext = *(const uint4*)(Wst + k0 + KSTEP);
#endif
            aNext = loadA(k0 + KSTEP);
        }

        const v16bf a = as_bf16(aReg);
        const unsigned short* Bc = Bsh + buf * LDSBUF;
#pragma unroll
        for (int nt = 0; nt < 4; ++nt) {
            // B 32x16: lane holds col N=lane%16; K = khalf*16 + [0..15] contiguous
            const unsigned short* p =
                Bc + (nt * 16 + (lane & 15)) * LDSCOL + khalf * 16;
            const uint4 lo = *(const uint4*)(p);
            const uint4 hi = *(const uint4*)(p + 8);
            v8u bu = {lo.x, lo.y, lo.z, lo.w, hi.x, hi.y, hi.z, hi.w};
            acc[nt] = __builtin_amdgcn_wmma_f32_16x16x32_bf16(
                false, a, false, as_bf16(bu), (short)0, acc[nt], false, false);
        }

        if (more) {
#ifdef USE_ASYNC_LDS
            wait_async0();                         // next buffer resident before barrier
#else
            *(uint4*)(Bsh + (buf ^ 1) * LDSBUF + ldsOff) = wNext;
#endif
            aReg = aNext;
        }
        __syncthreads();
    }

    // ---- C/D store: VGPR i -> M = i + khalf*8, N = lane%16 (ISA 7.12.2)
    const int colB = n0 + (lane & 15);
    const int mB   = m0 + khalf * 8;
#pragma unroll
    for (int i = 0; i < 8; ++i) {
        float* crow = C + (size_t)(mB + i) * N + colB;
        crow[0]  = acc[0][i];
        crow[16] = acc[1][i];
        crow[32] = acc[2][i];
        crow[48] = acc[3][i];
    }
}

__global__ __launch_bounds__(256) void wmma_gemm(
    const unsigned short* __restrict__ A, const unsigned short* __restrict__ W,
    float* __restrict__ C, int M, int N, int K)
{
    __shared__ __align__(16) unsigned short Bsh[2 * LDSBUF];
    gemm_tile(A, W, C, M, N, K, Bsh);
}

// Two independent GEMMs in one launch (layer-1 gi & gh), selected by blockIdx.y
__global__ __launch_bounds__(256) void wmma_gemm_dual(
    const unsigned short* __restrict__ A0, const unsigned short* __restrict__ W0,
    float* __restrict__ C0,
    const unsigned short* __restrict__ A1, const unsigned short* __restrict__ W1,
    float* __restrict__ C1,
    int M, int N, int K)
{
    __shared__ __align__(16) unsigned short Bsh[2 * LDSBUF];
    if (blockIdx.y == 0) gemm_tile(A0, W0, C0, M, N, K, Bsh);
    else                 gemm_tile(A1, W1, C1, M, N, K, Bsh);
}

// ---------------------------------------------------------------------------
// Layer-0 gates: gi gathered from precomputed embW0 (gather/linear commute).
// ---------------------------------------------------------------------------
__global__ void gate_l0(const int* __restrict__ input,       // [B,T]
                        int t,
                        const float* __restrict__ embW,      // [V, 3H]
                        const float* __restrict__ bi,        // [3H]
                        const float* __restrict__ bh,        // [3H]
                        const float* __restrict__ gh,        // [B, 3H]
                        const float* __restrict__ hprev,     // [B, H] f32
                        float* __restrict__ hout,            // [B, H] f32
                        unsigned short* __restrict__ hbf)    // [B, H] bf16
{
    int i = blockIdx.x * blockDim.x + threadIdx.x;
    if (i >= NB * NH) return;
    int b = i / NH, j = i % NH;
    int row = input[b * NT + t];
    const float* e = embW + (size_t)row * NG;
    const float* g = gh + (size_t)b * NG;

    float ir  = e[j]          + bi[j];
    float iz  = e[NH + j]     + bi[NH + j];
    float inn = e[2 * NH + j] + bi[2 * NH + j];
    float hr  = g[j]          + bh[j];
    float hz  = g[NH + j]     + bh[NH + j];
    float hn  = g[2 * NH + j] + bh[2 * NH + j];

    float r = sigmoidf(ir + hr);
    float z = sigmoidf(iz + hz);
    float n = tanhf(inn + r * hn);
    float h = (1.0f - z) * n + z * hprev[i];

    hout[i] = h;
    hbf[i]  = f32_to_bf16(h);
}

// ---------------------------------------------------------------------------
// Layer-1 gates
// ---------------------------------------------------------------------------
__global__ void gate_l1(const float* __restrict__ gi,        // [B, 3H]
                        const float* __restrict__ gh,        // [B, 3H]
                        const float* __restrict__ bi,        // [3H]
                        const float* __restrict__ bh,        // [3H]
                        const float* __restrict__ hprev,     // [B, H] f32
                        float* __restrict__ hout,            // [B, H] f32
                        unsigned short* __restrict__ hbf)    // [B, H] bf16
{
    int i = blockIdx.x * blockDim.x + threadIdx.x;
    if (i >= NB * NH) return;
    int b = i / NH, j = i % NH;
    const float* gi_b = gi + (size_t)b * NG;
    const float* gh_b = gh + (size_t)b * NG;

    float ir  = gi_b[j]          + bi[j];
    float iz  = gi_b[NH + j]     + bi[NH + j];
    float inn = gi_b[2 * NH + j] + bi[2 * NH + j];
    float hr  = gh_b[j]          + bh[j];
    float hz  = gh_b[NH + j]     + bh[NH + j];
    float hn  = gh_b[2 * NH + j] + bh[2 * NH + j];

    float r = sigmoidf(ir + hr);
    float z = sigmoidf(iz + hz);
    float n = tanhf(inn + r * hn);
    float h = (1.0f - z) * n + z * hprev[i];

    hout[i] = h;
    hbf[i]  = f32_to_bf16(h);
}

// ---------------------------------------------------------------------------
// Host-side orchestration
// ---------------------------------------------------------------------------
extern "C" void kernel_launch(void* const* d_in, const int* in_sizes, int n_in,
                              void* d_out, int out_size, void* d_ws, size_t ws_size,
                              hipStream_t stream) {
    const int*   input = (const int*)  d_in[0];   // [B,T]
    const float* enc   = (const float*)d_in[1];   // [L,B,H]
    const float* emb   = (const float*)d_in[2];   // [V,E]
    const float* w_ih  = (const float*)d_in[3];   // [L,3H,E]
    const float* w_hh  = (const float*)d_in[4];   // [L,3H,H]
    const float* b_ih  = (const float*)d_in[5];   // [L,3H]
    const float* b_hh  = (const float*)d_in[6];   // [L,3H]
    float* out = (float*)d_out;                   // states [T,L,B,H] ++ h_final [L,B,H]

    // ---- workspace carve-up (bf16 stored as ushort bit patterns)
    char* ws = (char*)d_ws;
    size_t off = 0;
    auto alloc = [&](size_t bytes) { char* p = ws + off; off += (bytes + 255) & ~size_t(255); return p; };
    unsigned short* emb_bf  = (unsigned short*)alloc((size_t)NV * NE * 2);
    unsigned short* wih0_bf = (unsigned short*)alloc((size_t)NG * NE * 2);
    unsigned short* whh0_bf = (unsigned short*)alloc((size_t)NG * NH * 2);
    unsigned short* wih1_bf = (unsigned short*)alloc((size_t)NG * NE * 2);
    unsigned short* whh1_bf = (unsigned short*)alloc((size_t)NG * NH * 2);
    float*          embW0   = (float*)alloc((size_t)NV * NG * 4);
    float*          gbuf0   = (float*)alloc((size_t)NB * NG * 4);
    float*          gbufA   = (float*)alloc((size_t)NB * NG * 4);
    float*          gbufB   = (float*)alloc((size_t)NB * NG * 4);
    unsigned short* h0_bf   = (unsigned short*)alloc((size_t)NB * NH * 2);
    unsigned short* h1_bf   = (unsigned short*)alloc((size_t)NB * NH * 2);

    auto cvt = [&](const float* s, unsigned short* d, int n) {
        cvt_f32_bf16<<<(n + 255) / 256, 256, 0, stream>>>(s, d, n);
    };

    // ---- one-time conversions (weights stay L2-resident: ~6 MB total)
    cvt(emb,                    emb_bf,  NV * NE);
    cvt(w_ih,                   wih0_bf, NG * NE);
    cvt(w_hh,                   whh0_bf, NG * NH);
    cvt(w_ih + (size_t)NG * NE, wih1_bf, NG * NE);
    cvt(w_hh + (size_t)NG * NH, whh1_bf, NG * NH);
    cvt(enc,                    h0_bf, NB * NH);
    cvt(enc + (size_t)NB * NH,  h1_bf, NB * NH);

    // ---- precompute embW0 = emb @ w_ih0^T  (V=128 rows; replaces a per-step GEMM)
    wmma_gemm<<<(NV / MBLK ? NV / MBLK : 1) * (NG / NBLK), 256, 0, stream>>>(
        emb_bf, wih0_bf, embW0, NV, NG, NE);

    const int gemmBlocks = (NB / MBLK) * (NG / NBLK);   // 4 * 24 = 96
    const int gateBlocks = (NB * NH + 255) / 256;
    const size_t stateStride = (size_t)NB * NH;

    for (int t = 0; t < NT; ++t) {
        float* st0 = out + ((size_t)t * NL + 0) * stateStride;
        float* st1 = out + ((size_t)t * NL + 1) * stateStride;
        const float* h0p = (t == 0) ? enc
                                    : out + ((size_t)(t - 1) * NL + 0) * stateStride;
        const float* h1p = (t == 0) ? enc + stateStride
                                    : out + ((size_t)(t - 1) * NL + 1) * stateStride;

        // layer 0: gh0 = h0 @ w_hh0^T ; gi0 gathered from embW0 inside gate
        wmma_gemm<<<gemmBlocks, 256, 0, stream>>>(h0_bf, whh0_bf, gbuf0, NB, NG, NH);
        gate_l0<<<gateBlocks, 256, 0, stream>>>(input, t, embW0, b_ih, b_hh,
                                                gbuf0, h0p, st0, h0_bf);

        // layer 1: gi1 = h0_new @ w_ih1^T  and  gh1 = h1 @ w_hh1^T in one launch
        wmma_gemm_dual<<<dim3(gemmBlocks, 2), 256, 0, stream>>>(
            h0_bf, wih1_bf, gbufA, h1_bf, whh1_bf, gbufB, NB, NG, NH);
        gate_l1<<<gateBlocks, 256, 0, stream>>>(gbufA, gbufB, b_ih + NG, b_hh + NG,
                                                h1p, st1, h1_bf);
    }

    // ---- h_final = states[T-1]  (graph-capturable d2d async copy)
    (void)hipMemcpyAsync(out + (size_t)NT * NL * stateStride,
                         out + (size_t)(NT - 1) * NL * stateStride,
                         (size_t)NL * stateStride * sizeof(float),
                         hipMemcpyDeviceToDevice, stream);
}
